// BaselineModel_47631187312790
// MI455X (gfx1250) — compile-verified, matrix operand order checked by reading, not run on hardware
//
#include <hip/hip_runtime.h>

#define TT 2048
#define BB 64
#define HH 64

typedef __attribute__((ext_vector_type(16))) _Float16 v16h;
typedef __attribute__((ext_vector_type(8)))  _Float16 v8h;
typedef __attribute__((ext_vector_type(8)))  float    v8f;

struct DirW { const _Float16* w; const float* b_ih; const float* b_hh; };
struct LayerArgs { DirW d[2]; };

// Convert and pack weights to f16, rows = 4H gate outputs, cols = [w_ih | w_hh] (K = in_dim+H)
__global__ void build_w_kernel(const float* __restrict__ w_ih,
                               const float* __restrict__ w_hh,
                               _Float16* __restrict__ out, int in_dim) {
  int K = in_dim + HH;
  int total = 4 * HH * K;
  for (int i = blockIdx.x * blockDim.x + threadIdx.x; i < total;
       i += gridDim.x * blockDim.x) {
    int n = i / K, k = i - n * K;
    float v = (k < in_dim) ? w_ih[n * in_dim + k] : w_hh[n * HH + (k - in_dim)];
    out[i] = (_Float16)v;
  }
}

__device__ __forceinline__ float fast_sig(float x) {
  return __builtin_amdgcn_rcpf(1.0f + __expf(-x)); // v_exp_f32 + v_rcp_f32
}
__device__ __forceinline__ float fast_tanh(float x) {
  // tanh(x) = 1 - 2/(1 + e^{2x})
  return 1.0f - 2.0f * __builtin_amdgcn_rcpf(1.0f + __expf(2.0f * x));
}

// Load one 16x16x32 f16 fragment (A or B layout; ISA 7.12.2: lane = M/N index,
// two contiguous 8xf16 K-chunks per lane) from a row-major [rows][stride] array.
template<typename P>
__device__ __forceinline__ v16h load_frag(const P* base, int row, int stride,
                                          int kbase, int grp) {
  const P* p = base + (size_t)row * stride + kbase + grp * 8;
  v8h lo = *(const v8h*)p;
  v8h hi = *(const v8h*)(p + 16);
  v16h f;
  #pragma unroll
  for (int e = 0; e < 8; ++e) { f[e] = lo[e]; f[8 + e] = hi[e]; }
  return f;
}

// One block = 32 batch rows of one direction of one layer; 8 waves; each wave
// owns one 16x16 cell tile (mt in 0..1, ht in 0..3) and its 4 gate tiles.
// The last KPIN K-slabs of the weight matrix (including the whole recurrent
// part) are pinned in VGPRs for the entire 2048-step loop; any leading slabs
// are staged once into LDS and fetched via the DS path each step.
template<int IN_DIM, bool FIRST, int KPIN>
__global__ __launch_bounds__(256)
void lstm_layer_kernel(const float* __restrict__ x0,        // FIRST: [T][B][IN_DIM] f32
                       const _Float16* __restrict__ y_in,   // !FIRST: [T][B][IN_DIM] f16
                       _Float16* __restrict__ y16_out,      // FIRST: [T][B][2H] f16
                       float* __restrict__ y32_out,         // !FIRST: [T][B][2H] f32
                       float* __restrict__ hc_out,          // d_out + T*B*H : h_n then c_n
                       LayerArgs args, int layer) {
  constexpr int K = IN_DIM + HH;           // concat [x_t | h]
  constexpr int NKK = K / 32;
  constexpr int KSTREAM = NKK - KPIN;      // leading slabs served from LDS
  constexpr int WS = (KSTREAM > 0) ? KSTREAM * 32 : 8; // LDS weight row stride
  constexpr int ROWS = 32;                 // batch rows per block
  const int dir   = blockIdx.y;            // 0 = fwd, 1 = rev
  const int rbase = blockIdx.x * ROWS;
  const int tid  = threadIdx.x;
  const int lane = tid & 31;
  const int wave = tid >> 5;
  const int grp  = lane >> 4;              // K half-group per ISA A/B layout
  const int n16  = lane & 15;              // M (A) / N (B,C,D) index within tile
  const int mt   = wave >> 2;              // 0..1: which 16-row slab
  const int ht   = wave & 3;               // 0..3: which 16-wide h column tile

  __shared__ alignas(16) _Float16 a_buf[ROWS * K]; // [32][K] f16, h at cols [IN_DIM, K)
  __shared__ alignas(16) _Float16 w_lds[(KSTREAM > 0) ? 4 * HH * WS : 8];

  const _Float16* __restrict__ wmat = args.d[dir].w;

  // per-lane bias scalars, one per gate (fused into the activation phase)
  float biasg[4];
  #pragma unroll
  for (int g = 0; g < 4; ++g) {
    int n = (g * 4 + ht) * 16 + n16;
    biasg[g] = args.d[dir].b_ih[n] + args.d[dir].b_hh[n];
  }

  // pin the tail K-slabs of this wave's 4 gate tiles in VGPRs (8 VGPRs each)
  v16h wreg[4][KPIN];
  #pragma unroll
  for (int kp = 0; kp < KPIN; ++kp)
    #pragma unroll
    for (int g = 0; g < 4; ++g)
      wreg[g][kp] = load_frag(wmat, (g * 4 + ht) * 16 + n16, K,
                              (KSTREAM + kp) * 32, grp);

  // stage the leading (x-projection) weight slabs into LDS once
  if constexpr (KSTREAM > 0) {
    for (int i = tid; i < 4 * HH * WS / 8; i += 256) {  // 8xf16 chunks
      int n = i / (WS / 8), c8 = i - n * (WS / 8);
      *(v8h*)(w_lds + n * WS + c8 * 8) =
          *(const v8h*)(wmat + (size_t)n * K + c8 * 8);
    }
  }

  // h0 = 0
  for (int i = tid; i < ROWS * HH; i += 256) {
    int r = i >> 6, c = i & (HH - 1);
    a_buf[r * K + IN_DIM + c] = (_Float16)0.0f;
  }
  __syncthreads();

  v8f creg; // c state lives in VGPRs for the whole sequence
  #pragma unroll
  for (int e = 0; e < 8; ++e) creg[e] = 0.0f;

  for (int s = 0; s < TT; ++s) {
    const int t = dir ? (TT - 1 - s) : s;

    // ---- stage x_t into LDS (f16) ----
    if (FIRST) {
      const float2* xp = (const float2*)(x0 + ((size_t)t * BB + rbase) * IN_DIM);
      #pragma unroll
      for (int it = 0; it < ROWS * IN_DIM / 512; ++it) {
        int i = it * 256 + tid;                 // pair index
        int r = i / (IN_DIM / 2), c2 = i - r * (IN_DIM / 2);
        float2 v = xp[i];
        a_buf[r * K + 2 * c2]     = (_Float16)v.x;
        a_buf[r * K + 2 * c2 + 1] = (_Float16)v.y;
      }
    } else {
      const v8h* xp = (const v8h*)(y_in + ((size_t)t * BB + rbase) * IN_DIM);
      #pragma unroll
      for (int it = 0; it < ROWS * IN_DIM / 2048; ++it) {
        int i = it * 256 + tid;                 // 8xf16 chunk index
        int r = i / (IN_DIM / 8), c8 = i - r * (IN_DIM / 8);
        *(v8h*)(a_buf + r * K + c8 * 8) = xp[i];
      }
    }
    // prefetch next timestep's x slab (global_prefetch_b8)
    {
      const int tn = dir ? (t - 1) : (t + 1);
      if (tn >= 0 && tn < TT && tid < 128) {
        if (FIRST)
          __builtin_prefetch(x0 + ((size_t)tn * BB + rbase) * IN_DIM + tid * 16, 0, 3);
        else
          __builtin_prefetch(y_in + ((size_t)tn * BB + rbase) * IN_DIM + tid * 32, 0, 3);
      }
    }
    __syncthreads(); // staging + previous h writes visible

    // ---- gates = [x_t|h] @ W^T via WMMA f16->f32; C starts at inline 0 ----
    v8f acc[4];
    #pragma unroll
    for (int g = 0; g < 4; ++g)
      #pragma unroll
      for (int e = 0; e < 8; ++e) acc[g][e] = 0.0f;

    // leading slabs (x-projection part; B fragments from LDS)
    #pragma unroll
    for (int kk = 0; kk < KSTREAM; ++kk) {
      v16h afrag = load_frag(a_buf, mt * 16 + n16, K, kk * 32, grp);
      #pragma unroll
      for (int g = 0; g < 4; ++g) {
        v16h bfrag = load_frag(w_lds, (g * 4 + ht) * 16 + n16, WS, kk * 32, grp);
        acc[g] = __builtin_amdgcn_wmma_f32_16x16x32_f16(
            false, afrag, false, bfrag, (short)0, acc[g], false, false);
      }
    }
    // pinned tail slabs (includes the entire latency-critical recurrent part)
    #pragma unroll
    for (int kp = 0; kp < KPIN; ++kp) {
      v16h afrag = load_frag(a_buf, mt * 16 + n16, K, (KSTREAM + kp) * 32, grp);
      #pragma unroll
      for (int g = 0; g < 4; ++g)
        acc[g] = __builtin_amdgcn_wmma_f32_16x16x32_f16(
            false, afrag, false, wreg[g][kp], (short)0, acc[g], false, false);
    }
    __syncthreads(); // all A-fragment LDS reads done before h region is overwritten

    // ---- LSTM cell update (bias fused here; dual-issues with v_exp chain) ----
    v8f hreg;
    #pragma unroll
    for (int j = 0; j < 8; ++j) {
      float iv = fast_sig(acc[0][j] + biasg[0]);
      float fv = fast_sig(acc[1][j] + biasg[1]);
      float gv = fast_tanh(acc[2][j] + biasg[2]);
      float ov = fast_sig(acc[3][j] + biasg[3]);
      float cn = fv * creg[j] + iv * gv;
      creg[j] = cn;
      hreg[j] = ov * fast_tanh(cn);
    }

    // ---- write h: LDS (next step's A) + global inter-layer buffer ----
    const int hcol = ht * 16 + n16;
    #pragma unroll
    for (int j = 0; j < 8; ++j) {
      int r = mt * 16 + grp * 8 + j; // C/D layout: VGPR j -> row j (+8 for lanes 16..31)
      _Float16 hv = (_Float16)hreg[j];
      a_buf[r * K + IN_DIM + hcol] = hv;
      size_t gidx = ((size_t)t * BB + rbase + r) * (2 * HH) + dir * HH + hcol;
      if (FIRST) y16_out[gidx] = hv;
      else       y32_out[gidx] = hreg[j];
    }
    if (s == TT - 1) { // final carry -> h_n / c_n
      int idx = layer * 2 + dir;
      #pragma unroll
      for (int j = 0; j < 8; ++j) {
        int r = mt * 16 + grp * 8 + j;
        size_t o = ((size_t)idx * BB + rbase + r) * HH + hcol;
        hc_out[o] = hreg[j];
        hc_out[(size_t)4 * BB * HH + o] = creg[j];
      }
    }
    // next iteration's post-staging barrier orders the h writes
  }
}

__global__ void finalize_selu_kernel(const float* __restrict__ y1,
                                     float* __restrict__ out, long n) {
  long i = (long)blockIdx.x * blockDim.x + threadIdx.x;
  if (i >= n) return;
  long tb = i >> 6;
  int h = (int)(i & 63);
  float v = y1[tb * 128 + h] + y1[tb * 128 + 64 + h];
  const float scale = 1.0507009873554805f;
  const float alpha = 1.6732632423543772f;
  out[i] = v > 0.0f ? scale * v : scale * alpha * (__expf(v) - 1.0f);
}

extern "C" void kernel_launch(void* const* d_in, const int* in_sizes, int n_in,
                              void* d_out, int out_size, void* d_ws, size_t ws_size,
                              hipStream_t stream) {
  (void)in_sizes; (void)n_in; (void)out_size; (void)ws_size;
  const float* x = (const float*)d_in[0];
  const float* w_ih[2][2]; const float* w_hh[2][2];
  const float* b_ih[2][2]; const float* b_hh[2][2];
  int p = 1;
  for (int l = 0; l < 2; ++l)
    for (int d = 0; d < 2; ++d) {
      w_ih[l][d] = (const float*)d_in[p++];
      w_hh[l][d] = (const float*)d_in[p++];
      b_ih[l][d] = (const float*)d_in[p++];
      b_hh[l][d] = (const float*)d_in[p++];
    }

  // workspace: y0 (layer0 out, f16 [T][B][128]) | y1 (layer1 out, f32 [T][B][128]) | f16 weights
  size_t y0_bytes = (size_t)TT * BB * 2 * HH * sizeof(_Float16); // 33.5 MB
  size_t y1_bytes = (size_t)TT * BB * 2 * HH * sizeof(float);    // 67.1 MB
  _Float16* y0 = (_Float16*)d_ws;
  float*    y1 = (float*)((char*)d_ws + y0_bytes);
  _Float16* wc = (_Float16*)((char*)d_ws + y0_bytes + y1_bytes);
  _Float16* wc0[2] = { wc, wc + 256 * 128 };
  _Float16* wc1[2] = { wc + 2 * 256 * 128, wc + 2 * 256 * 128 + 256 * 192 };

  for (int d = 0; d < 2; ++d) {
    build_w_kernel<<<64, 256, 0, stream>>>(w_ih[0][d], w_hh[0][d], wc0[d], 64);
    build_w_kernel<<<96, 256, 0, stream>>>(w_ih[1][d], w_hh[1][d], wc1[d], 128);
  }

  float* hc = (float*)d_out + (size_t)TT * BB * HH;

  LayerArgs a0, a1;
  for (int d = 0; d < 2; ++d) {
    a0.d[d] = { wc0[d], b_ih[0][d], b_hh[0][d] };
    a1.d[d] = { wc1[d], b_ih[1][d], b_hh[1][d] };
  }

  // layer 0: all 4 K-slabs pinned in VGPRs; layer 1: pin last 4 (incl. all
  // recurrent weights), serve first 2 x-projection slabs from LDS.
  lstm_layer_kernel<64, true, 4><<<dim3(2, 2), 256, 0, stream>>>(
      x, nullptr, y0, nullptr, hc, a0, 0);
  lstm_layer_kernel<128, false, 4><<<dim3(2, 2), 256, 0, stream>>>(
      nullptr, y0, nullptr, y1, hc, a1, 1);

  long n = (long)TT * BB * HH;
  finalize_selu_kernel<<<(unsigned)((n + 255) / 256), 256, 0, stream>>>(
      y1, (float*)d_out, n);
}
